// SwinTransformer_3d_63634235457933
// MI455X (gfx1250) — matrix operand order repaired and checked by reading, not compiled, not run on hardware
//
#include <hip/hip_runtime.h>
#include <hip/hip_bf16.h>
#include <cstdint>

// ---------------------------------------------------------------------------
// Types for CDNA5 WMMA (wave32): v_wmma_f32_16x16x32_bf16
// ---------------------------------------------------------------------------
typedef __attribute__((ext_vector_type(16))) __bf16 v16bf;
typedef __attribute__((ext_vector_type(8)))  __bf16 v8bf;
typedef __attribute__((ext_vector_type(8)))  float  v8f;

__device__ __forceinline__ v16bf cat8(v8bf lo, v8bf hi) {
  return __builtin_shufflevector(lo, hi, 0, 1, 2, 3, 4, 5, 6, 7,
                                 8, 9, 10, 11, 12, 13, 14, 15);
}

// Problem constants
#define BB 4
#define TT 16
#define HH 32
#define WW 32
#define CC 256
#define HEADS 8
#define HD 32
#define NWIN 64            // tokens per window (4^3)
#define BW_TOT 1024        // total windows (B * 256)
#define MROWS 65536        // B*T*H*W
#define MLPD 1024
#define QK_SCALE 0.17677669529663687f  // 32^-0.5

// window-row -> shifted token index (used for both gather in LN1 and scatter in proj)
__device__ __forceinline__ size_t shifted_token(int m) {
  int bw = m >> 6, n = m & 63;
  int bb = bw >> 8, widx = bw & 255;
  int tp = ((widx >> 6) << 2) + (n >> 4);
  int hp = (((widx >> 3) & 7) << 2) + ((n >> 2) & 3);
  int wp = ((widx & 7) << 2) + (n & 3);
  int t  = (tp + 2) & 15;
  int h2 = (hp + 2) & 31;
  int w2 = (wp + 2) & 31;
  return (((size_t)bb * TT + t) * HH + h2) * WW + w2;
}

// ---------------------------------------------------------------------------
// Shared WMMA GEMM core: block tile 128(M) x 64(N), K in steps of 32.
// 256 threads = 8 waves in a 4(M) x 2(N) grid; each wave owns a 32x32 subtile
// = 2x2 WMMA 16x16 accumulators.  A,B are bf16 row-major.  B tile is staged
// transposed in LDS so both fragment loads are contiguous b128s.
// ---------------------------------------------------------------------------
__device__ __forceinline__ void gemm_core(const __bf16* __restrict__ A, int lda,
                                          const __bf16* __restrict__ B, int ldb,
                                          int m0, int n0, int K,
                                          __bf16* As, __bf16* BsT,
                                          v8f acc[2][2]) {
  const int tid  = threadIdx.x;
  const int wave = tid >> 5, lane = tid & 31;
  const int wm = wave & 3, wn = wave >> 2;
  const int hl = lane >> 4, lm = lane & 15;

  v8f zero = {0.f, 0.f, 0.f, 0.f, 0.f, 0.f, 0.f, 0.f};
  acc[0][0] = zero; acc[0][1] = zero; acc[1][0] = zero; acc[1][1] = zero;

  for (int k0 = 0; k0 < K; k0 += 32) {
    // --- stage A tile: 128x32 bf16 (each thread: two contiguous v8bf) ---
#pragma unroll
    for (int p = 0; p < 2; ++p) {
      int e = tid * 8 + p * 2048;
      int r = e >> 5, c = e & 31;
      const __bf16* src = &A[(size_t)(m0 + r) * lda + k0 + c];
      *(v8bf*)&As[e] = *(const v8bf*)src;
      if (k0 + 32 < K) __builtin_prefetch(src + 32, 0, 1);  // global_prefetch_b8
    }
    // --- stage B tile transposed: (32xK)x(64 N) -> BsT[64][32] ---
    {
      int e = tid * 8;
      int r = e >> 6, c = e & 63;
      const __bf16* src = &B[(size_t)(k0 + r) * ldb + n0 + c];
      v8bf t = *(const v8bf*)src;
      if (k0 + 32 < K) __builtin_prefetch(src + (size_t)32 * ldb, 0, 1);
#pragma unroll
      for (int i = 0; i < 8; ++i) BsT[(c + i) * 32 + r] = t[i];
    }
    __syncthreads();

    // --- fragments (CDNA5 16-bit A layout: chunks [8*hl,+8) and [16+8*hl,+8);
    //     B layout: column = lane%16, same K chunks [8*hl,+8) and [16+8*hl,+8)) ---
    v16bf af[2], bfr[2];
#pragma unroll
    for (int mi = 0; mi < 2; ++mi) {
      const __bf16* ap = &As[(wm * 32 + mi * 16 + lm) * 32];
      af[mi] = cat8(*(const v8bf*)&ap[8 * hl], *(const v8bf*)&ap[16 + 8 * hl]);
    }
#pragma unroll
    for (int ni = 0; ni < 2; ++ni) {
      const __bf16* bp = &BsT[(wn * 32 + ni * 16 + lm) * 32];
      bfr[ni] = cat8(*(const v8bf*)&bp[8 * hl], *(const v8bf*)&bp[16 + 8 * hl]);
    }
#pragma unroll
    for (int mi = 0; mi < 2; ++mi)
#pragma unroll
      for (int ni = 0; ni < 2; ++ni)
        acc[mi][ni] = __builtin_amdgcn_wmma_f32_16x16x32_bf16(
            false, af[mi], false, bfr[ni], (short)0, acc[mi][ni], false, false);
    __syncthreads();
  }
}

// Epilogue index helpers: D-tile lane holds col = lane%16, rows r + 8*(lane/16)
// Variadic so the body may contain top-level commas.
#define EPILOGUE_LOOP(...)                                                    \
  {                                                                           \
    const int tid = threadIdx.x;                                              \
    const int wave = tid >> 5, lane = tid & 31;                               \
    const int wm = wave & 3, wn = wave >> 2;                                  \
    const int hl = lane >> 4, lm = lane & 15;                                 \
    _Pragma("unroll") for (int mi = 0; mi < 2; ++mi)                          \
    _Pragma("unroll") for (int ni = 0; ni < 2; ++ni)                          \
    _Pragma("unroll") for (int r = 0; r < 8; ++r) {                           \
      int m = m0 + wm * 32 + mi * 16 + r + 8 * hl;                            \
      int n = n0 + wn * 32 + ni * 16 + lm;                                    \
      float val = acc[mi][ni][r];                                             \
      __VA_ARGS__                                                             \
    }                                                                         \
  }

// ---------------------------------------------------------------------------
// K0: weight fp32->bf16 conversion + dense rel-pos bias gather
// ---------------------------------------------------------------------------
__global__ void prep_kernel(const float* __restrict__ qkv_w,
                            const float* __restrict__ proj_w,
                            const float* __restrict__ mlp_w1,
                            const float* __restrict__ mlp_w2,
                            const float* __restrict__ table,
                            const int* __restrict__ rel_index,
                            __bf16* wq, __bf16* wp, __bf16* w1, __bf16* w2,
                            float* bias_full) {
  int i = blockIdx.x * blockDim.x + threadIdx.x;
  int stride = gridDim.x * blockDim.x;
  for (int e = i; e < CC * 3 * CC; e += stride) wq[e] = (__bf16)qkv_w[e];
  for (int e = i; e < CC * CC; e += stride) wp[e] = (__bf16)proj_w[e];
  for (int e = i; e < CC * MLPD; e += stride) w1[e] = (__bf16)mlp_w1[e];
  for (int e = i; e < MLPD * CC; e += stride) w2[e] = (__bf16)mlp_w2[e];
  for (int e = i; e < HEADS * NWIN * NWIN; e += stride) {
    int h = e >> 12, ij = e & 4095;
    bias_full[e] = table[rel_index[ij] * HEADS + h];
  }
}

// ---------------------------------------------------------------------------
// K1: LN1 + roll(-2,-2,-2) + window partition -> bf16 Hwin[65536][256]
// ---------------------------------------------------------------------------
__global__ __launch_bounds__(256) void ln1_kernel(const float* __restrict__ x,
                                                  const float* __restrict__ g,
                                                  const float* __restrict__ b,
                                                  __bf16* __restrict__ Hwin) {
  __shared__ float red[256];
  int m = blockIdx.x, c = threadIdx.x;
  size_t tok = shifted_token(m);
  float v = x[tok * CC + c];
  red[c] = v; __syncthreads();
  for (int s = 128; s > 0; s >>= 1) { if (c < s) red[c] += red[c + s]; __syncthreads(); }
  float mu = red[0] * (1.0f / 256.0f);
  __syncthreads();
  float d = v - mu;
  red[c] = d * d; __syncthreads();
  for (int s = 128; s > 0; s >>= 1) { if (c < s) red[c] += red[c + s]; __syncthreads(); }
  float var = red[0] * (1.0f / 256.0f);
  Hwin[(size_t)m * CC + c] = (__bf16)(d * rsqrtf(var + 1e-5f) * g[c] + b[c]);
}

// ---------------------------------------------------------------------------
// K2: QKV GEMM (65536x256 @ 256x768) -> scaled-q / k / v, per-(window,head)
// ---------------------------------------------------------------------------
__global__ __launch_bounds__(256) void qkv_gemm_kernel(
    const __bf16* __restrict__ Hwin, const __bf16* __restrict__ Wq,
    const float* __restrict__ qkv_b, __bf16* __restrict__ qp,
    __bf16* __restrict__ kp, __bf16* __restrict__ vp) {
  __shared__ __bf16 As[128 * 32];
  __shared__ __bf16 BsT[64 * 32];
  int m0 = blockIdx.x * 128, n0 = blockIdx.y * 64;
  v8f acc[2][2];
  gemm_core(Hwin, CC, Wq, 3 * CC, m0, n0, CC, As, BsT, acc);
  EPILOGUE_LOOP(
    val += qkv_b[n];
    int which = n >> 8;
    int rem = n & 255;
    int hh = rem >> 5;
    int d = rem & 31;
    size_t idx = (((size_t)(m >> 6) * HEADS + hh) * NWIN + (m & 63)) * HD + d;
    if (which == 0)      qp[idx] = (__bf16)(val * QK_SCALE);
    else if (which == 1) kp[idx] = (__bf16)val;
    else                 vp[idx] = (__bf16)val;
  )
}

// ---------------------------------------------------------------------------
// K3: attention per (window, head): S=q*k^T (WMMA), +bias+mask, softmax,
//     O=P*V (WMMA).  128 threads = 4 waves; each wave owns 16 rows.
// ---------------------------------------------------------------------------
__global__ __launch_bounds__(128) void attn_kernel(
    const __bf16* __restrict__ qp, const __bf16* __restrict__ kp,
    const __bf16* __restrict__ vp, const float* __restrict__ bias_full,
    const float* __restrict__ mask, __bf16* __restrict__ o) {
  __shared__ float  S[64 * 64];
  __shared__ __bf16 P[64 * 64];
  __shared__ __bf16 Vt[32 * 64];  // Vt[d][j] = v[j][d]

  int blk = blockIdx.x;            // bw*8 + head
  int bw = blk >> 3, h = blk & 7;
  const __bf16* q = qp + (size_t)blk * NWIN * HD;
  const __bf16* k = kp + (size_t)blk * NWIN * HD;
  const __bf16* v = vp + (size_t)blk * NWIN * HD;

  int tid = threadIdx.x, wave = tid >> 5, lane = tid & 31;
  int hl = lane >> 4, lm = lane & 15;
  int i0 = wave * 16;

  // stage V transposed into LDS
  {
    int j = tid & 63, dh = tid >> 6;
    v8bf a = *(const v8bf*)&v[j * HD + dh * 16];
    v8bf bv = *(const v8bf*)&v[j * HD + dh * 16 + 8];
#pragma unroll
    for (int e = 0; e < 8; ++e) Vt[(dh * 16 + e) * 64 + j] = a[e];
#pragma unroll
    for (int e = 0; e < 8; ++e) Vt[(dh * 16 + 8 + e) * 64 + j] = bv[e];
  }

  // phase 1: S rows [i0,i0+16) = q @ k^T  (K = head_dim = 32: single WMMA/tile)
  v16bf qf;
  {
    const __bf16* ap = &q[(i0 + lm) * HD];
    qf = cat8(*(const v8bf*)&ap[8 * hl], *(const v8bf*)&ap[16 + 8 * hl]);
  }
#pragma unroll
  for (int jt = 0; jt < 4; ++jt) {
    const __bf16* bp = &k[(jt * 16 + lm) * HD];
    v16bf kf = cat8(*(const v8bf*)&bp[8 * hl], *(const v8bf*)&bp[16 + 8 * hl]);
    v8f s = {0.f, 0.f, 0.f, 0.f, 0.f, 0.f, 0.f, 0.f};
    s = __builtin_amdgcn_wmma_f32_16x16x32_bf16(false, qf, false, kf, (short)0,
                                                s, false, false);
#pragma unroll
    for (int r = 0; r < 8; ++r)
      S[(i0 + r + 8 * hl) * 64 + jt * 16 + lm] = s[r];
  }
  __syncthreads();

  // phase 2: softmax with rel-pos bias + shifted-window mask
  if (tid < 64) {
    int i = tid, widx = bw & 255;
    const float* bi = bias_full + h * 4096 + i * 64;
    const float* mk = mask + (size_t)widx * 4096 + i * 64;
    float mx = -1e30f;
    for (int j = 0; j < 64; ++j) {
      float vv = S[i * 64 + j] + bi[j] + mk[j];
      S[i * 64 + j] = vv;
      mx = fmaxf(mx, vv);
    }
    float sum = 0.f;
    for (int j = 0; j < 64; ++j) {
      float e = __expf(S[i * 64 + j] - mx);
      S[i * 64 + j] = e;
      sum += e;
    }
    float rs = 1.0f / sum;
    for (int j = 0; j < 64; ++j) P[i * 64 + j] = (__bf16)(S[i * 64 + j] * rs);
  }
  __syncthreads();

  // phase 3: O rows [i0,i0+16) = P @ V   (K = 64 -> two WMMA K-steps)
  v8f oz = {0.f, 0.f, 0.f, 0.f, 0.f, 0.f, 0.f, 0.f};
  v8f oacc[2] = {oz, oz};
#pragma unroll
  for (int ks = 0; ks < 64; ks += 32) {
    const __bf16* ap = &P[(i0 + lm) * 64 + ks];
    v16bf pf = cat8(*(const v8bf*)&ap[8 * hl], *(const v8bf*)&ap[16 + 8 * hl]);
#pragma unroll
    for (int dt = 0; dt < 2; ++dt) {
      const __bf16* bp = &Vt[(dt * 16 + lm) * 64 + ks];
      v16bf vf = cat8(*(const v8bf*)&bp[8 * hl], *(const v8bf*)&bp[16 + 8 * hl]);
      oacc[dt] = __builtin_amdgcn_wmma_f32_16x16x32_bf16(
          false, pf, false, vf, (short)0, oacc[dt], false, false);
    }
  }
#pragma unroll
  for (int dt = 0; dt < 2; ++dt)
#pragma unroll
    for (int r = 0; r < 8; ++r) {
      int i = i0 + r + 8 * hl, d = dt * 16 + lm;
      o[((size_t)bw * NWIN + i) * CC + h * HD + d] = (__bf16)oacc[dt][r];
    }
}

// ---------------------------------------------------------------------------
// K4: proj GEMM + window-reverse + roll(+2) + residual -> fp32 out (token order)
// ---------------------------------------------------------------------------
__global__ __launch_bounds__(256) void proj_gemm_kernel(
    const __bf16* __restrict__ o, const __bf16* __restrict__ Wp,
    const float* __restrict__ proj_b, const float* __restrict__ x,
    float* __restrict__ out) {
  __shared__ __bf16 As[128 * 32];
  __shared__ __bf16 BsT[64 * 32];
  int m0 = blockIdx.x * 128, n0 = blockIdx.y * 64;
  v8f acc[2][2];
  gemm_core(o, CC, Wp, CC, m0, n0, CC, As, BsT, acc);
  EPILOGUE_LOOP(
    size_t tok = shifted_token(m);
    out[tok * CC + n] = x[tok * CC + n] + val + proj_b[n];
  )
}

// ---------------------------------------------------------------------------
// K5: LN2 on out (token order) -> bf16 y
// ---------------------------------------------------------------------------
__global__ __launch_bounds__(256) void ln2_kernel(const float* __restrict__ xin,
                                                  const float* __restrict__ g,
                                                  const float* __restrict__ b,
                                                  __bf16* __restrict__ y) {
  __shared__ float red[256];
  int m = blockIdx.x, c = threadIdx.x;
  float v = xin[(size_t)m * CC + c];
  red[c] = v; __syncthreads();
  for (int s = 128; s > 0; s >>= 1) { if (c < s) red[c] += red[c + s]; __syncthreads(); }
  float mu = red[0] * (1.0f / 256.0f);
  __syncthreads();
  float d = v - mu;
  red[c] = d * d; __syncthreads();
  for (int s = 128; s > 0; s >>= 1) { if (c < s) red[c] += red[c + s]; __syncthreads(); }
  float var = red[0] * (1.0f / 256.0f);
  y[(size_t)m * CC + c] = (__bf16)(d * rsqrtf(var + 1e-6f) * g[c] + b[c]);
}

// ---------------------------------------------------------------------------
// K6: MLP1 GEMM (65536x256 @ 256x1024) + exact GELU -> bf16 h1
// ---------------------------------------------------------------------------
__global__ __launch_bounds__(256) void mlp1_gemm_kernel(
    const __bf16* __restrict__ y, const __bf16* __restrict__ W1,
    const float* __restrict__ b1, __bf16* __restrict__ h1) {
  __shared__ __bf16 As[128 * 32];
  __shared__ __bf16 BsT[64 * 32];
  int m0 = blockIdx.x * 128, n0 = blockIdx.y * 64;
  v8f acc[2][2];
  gemm_core(y, CC, W1, MLPD, m0, n0, CC, As, BsT, acc);
  EPILOGUE_LOOP(
    float z = val + b1[n];
    float ge = 0.5f * z * (1.0f + erff(z * 0.7071067811865475f));
    h1[(size_t)m * MLPD + n] = (__bf16)ge;
  )
}

// ---------------------------------------------------------------------------
// K7: MLP2 GEMM (65536x1024 @ 1024x256) + bias + residual into out
// ---------------------------------------------------------------------------
__global__ __launch_bounds__(256) void mlp2_gemm_kernel(
    const __bf16* __restrict__ h1, const __bf16* __restrict__ W2,
    const float* __restrict__ b2, float* __restrict__ out) {
  __shared__ __bf16 As[128 * 32];
  __shared__ __bf16 BsT[64 * 32];
  int m0 = blockIdx.x * 128, n0 = blockIdx.y * 64;
  v8f acc[2][2];
  gemm_core(h1, MLPD, W2, CC, m0, n0, MLPD, As, BsT, acc);
  EPILOGUE_LOOP(
    size_t idx = (size_t)m * CC + n;
    out[idx] = out[idx] + val + b2[n];
  )
}

// ---------------------------------------------------------------------------
// Launch
// ---------------------------------------------------------------------------
extern "C" void kernel_launch(void* const* d_in, const int* in_sizes, int n_in,
                              void* d_out, int out_size, void* d_ws,
                              size_t ws_size, hipStream_t stream) {
  (void)in_sizes; (void)n_in; (void)out_size; (void)ws_size;
  const float* x        = (const float*)d_in[0];
  const float* n1g      = (const float*)d_in[1];
  const float* n1b      = (const float*)d_in[2];
  const float* qkv_w    = (const float*)d_in[3];
  const float* qkv_b    = (const float*)d_in[4];
  const float* proj_w   = (const float*)d_in[5];
  const float* proj_b   = (const float*)d_in[6];
  const float* rel_tab  = (const float*)d_in[7];
  const float* n2g      = (const float*)d_in[8];
  const float* n2b      = (const float*)d_in[9];
  const float* mlp_w1   = (const float*)d_in[10];
  const float* mlp_b1   = (const float*)d_in[11];
  const float* mlp_w2   = (const float*)d_in[12];
  const float* mlp_b2   = (const float*)d_in[13];
  const int*   rel_idx  = (const int*)d_in[14];
  const float* attn_msk = (const float*)d_in[15];
  float* out = (float*)d_out;

  // Workspace layout (reuse regions; total ~164 MB)
  char* ws = (char*)d_ws;
  __bf16* wq = (__bf16*)ws;                 // 196608
  __bf16* wp = wq + 196608;                 // 65536
  __bf16* w1 = wp + 65536;                  // 262144
  __bf16* w2 = w1 + 262144;                 // 262144
  float* bias_full = (float*)(ws + 2ull * 1024 * 1024);        // 8*64*64 f32
  __bf16* buf1 = (__bf16*)(ws + 4ull * 1024 * 1024);           // 32MB: Hwin -> o -> y
  __bf16* qp = (__bf16*)(ws + 36ull * 1024 * 1024);            // 32MB
  __bf16* kp = (__bf16*)(ws + 68ull * 1024 * 1024);            // 32MB
  __bf16* vp = (__bf16*)(ws + 100ull * 1024 * 1024);           // 32MB
  __bf16* h1 = qp;                                             // 128MB reuse (36..164MB)

  prep_kernel<<<512, 256, 0, stream>>>(qkv_w, proj_w, mlp_w1, mlp_w2, rel_tab,
                                       rel_idx, wq, wp, w1, w2, bias_full);
  ln1_kernel<<<MROWS, 256, 0, stream>>>(x, n1g, n1b, buf1);
  qkv_gemm_kernel<<<dim3(MROWS / 128, 12), 256, 0, stream>>>(buf1, wq, qkv_b,
                                                             qp, kp, vp);
  attn_kernel<<<BW_TOT * HEADS, 128, 0, stream>>>(qp, kp, vp, bias_full,
                                                  attn_msk, buf1);
  proj_gemm_kernel<<<dim3(MROWS / 128, 4), 256, 0, stream>>>(buf1, wp, proj_b,
                                                             x, out);
  ln2_kernel<<<MROWS, 256, 0, stream>>>(out, n2g, n2b, buf1);
  mlp1_gemm_kernel<<<dim3(MROWS / 128, 16), 256, 0, stream>>>(buf1, w1, mlp_b1,
                                                              h1);
  mlp2_gemm_kernel<<<dim3(MROWS / 128, 4), 256, 0, stream>>>(h1, w2, mlp_b2,
                                                             out);
}
